// EPD_with_sampling_25769804176
// MI455X (gfx1250) — matrix operand order repaired and checked by reading, not compiled, not run on hardware
//
#include <hip/hip_runtime.h>

typedef _Float16 f16;
typedef _Float16 v16h __attribute__((ext_vector_type(16)));
typedef float    v8f  __attribute__((ext_vector_type(8)));

union F16Frag { v16h v; uint4 q[2]; };

#define HDIM 128
constexpr int GNUM    = 16;   // graphs
constexpr int REPEATS = 3;

// Async global->LDS staging (GLOBAL_LOAD_ASYNC_TO_LDS_B128, ASYNCcnt) —
// verified to assemble for gfx1250 in round 2.
#define USE_ASYNC 1

typedef __attribute__((address_space(3))) f16 lds_f16;

__device__ __forceinline__ void async_copy_b128(f16* lds_dst, const f16* gsrc) {
#if USE_ASYNC
  asm volatile("global_load_async_to_lds_b128 %0, %1, off"
               :
               : "v"((lds_f16*)lds_dst), "v"(gsrc)
               : "memory");
#else
  *(uint4*)lds_dst = *(const uint4*)gsrc;
#endif
}
__device__ __forceinline__ void wait_async0() {
#if USE_ASYNC
  asm volatile("s_wait_asynccnt 0" ::: "memory");
#endif
}

// ---------------------------------------------------------------------------
// WMMA fragment loaders (ISA 7.12.2 layouts, wave32)
// ---------------------------------------------------------------------------
__device__ __forceinline__ v16h load_a_frag_lds(const f16* A, int stride_h, int kbase, int lane) {
  int row  = lane & 15;
  int koff = kbase + ((lane >> 4) << 3);
  const f16* p = A + row * stride_h + koff;
  F16Frag r;
  r.q[0] = *(const uint4*)(p);        // K = koff .. koff+7
  r.q[1] = *(const uint4*)(p + 16);   // K = koff+16 .. koff+23
  return r.v;
}

__device__ __forceinline__ v16h load_b_frag(const f16* Wt, int ldk, int n0, int kbase, int lane) {
  int n    = n0 + (lane & 15);
  int koff = kbase + ((lane >> 4) << 3);
  const f16* p = Wt + (size_t)n * ldk + koff;
  F16Frag r;
  r.q[0] = *(const uint4*)(p);
  r.q[1] = *(const uint4*)(p + 16);
  return r.v;
}

__device__ __forceinline__ v8f wmma_step(v16h a, v16h b, v8f c) {
  return __builtin_amdgcn_wmma_f32_16x16x32_f16(false, a, false, b, (short)0, c, false, false);
}

// ---------------------------------------------------------------------------
// Small utility kernels
// ---------------------------------------------------------------------------
__global__ void zero_f32(float* __restrict__ p, int n) {
  int i = blockIdx.x * blockDim.x + threadIdx.x;
  int stride = gridDim.x * blockDim.x;
  for (; i < n; i += stride) p[i] = 0.0f;
}

__global__ void make_wt(const float* __restrict__ W, f16* __restrict__ Wt, int K) {
  int idx = blockIdx.x * blockDim.x + threadIdx.x;
  if (idx >= K * HDIM) return;
  int k = idx >> 7, n = idx & 127;
  Wt[(size_t)n * K + k] = (f16)W[idx];
}

__global__ void degree_kernel(const int* __restrict__ dst, float* __restrict__ deg, int E) {
  int i = blockIdx.x * blockDim.x + threadIdx.x;
  if (i < E) atomicAdd(&deg[dst[i]], 1.0f);
}

__global__ void finalize_deg(float* __restrict__ deg, int N) {
  int i = blockIdx.x * blockDim.x + threadIdx.x;
  if (i < N) deg[i] = 1.0f / fmaxf(deg[i], 1.0f);
}

__global__ void node_counts(const int* __restrict__ batch, const float* __restrict__ x_mask,
                            float* __restrict__ n_per_g, float* __restrict__ bc_per_g, int N) {
  int i = blockIdx.x * blockDim.x + threadIdx.x;
  if (i >= N) return;
  int g = batch[i];
  atomicAdd(&n_per_g[g], 1.0f);
  atomicAdd(&bc_per_g[g], x_mask[(size_t)i * 4 + 3]);
}

__global__ __launch_bounds__(HDIM) void enc_kernel(
    const float* __restrict__ x, const float* __restrict__ x_mask,
    const float* __restrict__ enc_W, const float* __restrict__ enc_b,
    const int* __restrict__ batch,
    float* __restrict__ h, f16* __restrict__ h_half,
    float* __restrict__ xg_sum, float* __restrict__ xbc_sum, int N) {
  int node = blockIdx.x;
  if (node >= N) return;
  __shared__ float in[12];
  int j = threadIdx.x;
  if (j < 8)       in[j] = x[(size_t)node * 8 + j];
  else if (j < 12) in[j] = x_mask[(size_t)node * 4 + (j - 8)];
  __syncthreads();
  float s = enc_b[j];
#pragma unroll
  for (int k = 0; k < 12; ++k) s = fmaf(in[k], enc_W[k * HDIM + j], s);
  s = fmaxf(s, 0.0f);
  h[(size_t)node * HDIM + j]      = s;
  h_half[(size_t)node * HDIM + j] = (f16)s;
  int g = batch[node];
  float bc = in[11];
  atomicAdd(&xg_sum[g * HDIM + j], s);
  atomicAdd(&xbc_sum[g * HDIM + j], s * bc);
}

__global__ void finalize_graphs(const float* __restrict__ xg_sum, const float* __restrict__ xbc_sum,
                                const float* __restrict__ npg, const float* __restrict__ bcg,
                                f16* __restrict__ xg_h, f16* __restrict__ xbc_h) {
  int i = blockIdx.x * blockDim.x + threadIdx.x;
  if (i >= GNUM * HDIM) return;
  int g = i >> 7;
  xg_h[i]  = (f16)(xg_sum[i]  / fmaxf(npg[g], 1.0f));
  xbc_h[i] = (f16)(xbc_sum[i] / fmaxf(bcg[g], 1.0f));
}

__global__ void finalize_xg(const float* __restrict__ xg_sum, const float* __restrict__ npg,
                            f16* __restrict__ xg_h) {
  int i = blockIdx.x * blockDim.x + threadIdx.x;
  if (i >= GNUM * HDIM) return;
  int g = i >> 7;
  xg_h[i] = (f16)(xg_sum[i] / fmaxf(npg[g], 1.0f));
}

// ---------------------------------------------------------------------------
// Message MLP: m = relu([h[src]|h[dst]|edge_attr] @ msg_W + b); agg += m/deg
// ---------------------------------------------------------------------------
constexpr int MSG_K     = 2 * HDIM;        // 256
constexpr int MSG_AS    = MSG_K + 8;       // halves; row = 528B (16B multiple)
constexpr int MSG_TILES = 8;               // 128 edges per block

__global__ __launch_bounds__(256) void msg_kernel(
    const f16* __restrict__ h_half, const float* __restrict__ edge_attr,
    const int* __restrict__ src, const int* __restrict__ dst,
    const float* __restrict__ inv_deg,
    const f16* __restrict__ Wt,            // [128][256] fp16
    const float* __restrict__ msg_W,       // [259][128] fp32 (rows 256..258)
    const float* __restrict__ msg_b,
    float* __restrict__ agg, int E) {
  __shared__ __align__(16) f16 Alds[2][16 * MSG_AS];
  __shared__ float ea[2][16 * 3];
  __shared__ int   dstl[2][16];
  __shared__ float idg[2][16];

  int tid  = threadIdx.x;
  int lane = tid & 31;
  int wave = tid >> 5;
  int n0   = wave * 16;

  v16h bfrag[8];
#pragma unroll
  for (int ks = 0; ks < 8; ++ks) bfrag[ks] = load_b_frag(Wt, MSG_K, n0, ks * 32, lane);

  int   nlane = n0 + (lane & 15);
  float wb = msg_b[nlane];
  float w0 = msg_W[256 * HDIM + nlane];
  float w1 = msg_W[257 * HDIM + nlane];
  float w2 = msg_W[258 * HDIM + nlane];

  int  ebase   = blockIdx.x * (16 * MSG_TILES);
  bool fullblk = (ebase + 16 * MSG_TILES) <= E;   // block-uniform

  auto stage = [&](int t, int buf) {
    int e0 = ebase + t * 16;
    int m = tid >> 4, c = tid & 15;               // c: 16-half chunk (32B)
    int e = e0 + m;
    f16* out = &Alds[buf][m * MSG_AS + c * 16];
    if (fullblk || e < E) {
      int node = (c < 8) ? src[e] : dst[e];
      int off  = (c & 7) * 16;
      const f16* p = h_half + (size_t)node * HDIM + off;
      async_copy_b128(out, p);
      async_copy_b128(out + 8, p + 8);
    } else {
      uint4 z{0, 0, 0, 0};
      *(uint4*)(out) = z;  *(uint4*)(out + 8) = z;
    }
    if (tid < 16) {
      int ee = e0 + tid;
      if (fullblk || ee < E) {
        int d = dst[ee];
        dstl[buf][tid] = d;
        idg[buf][tid]  = inv_deg[d];
        ea[buf][tid * 3 + 0] = edge_attr[(size_t)ee * 3 + 0];
        ea[buf][tid * 3 + 1] = edge_attr[(size_t)ee * 3 + 1];
        ea[buf][tid * 3 + 2] = edge_attr[(size_t)ee * 3 + 2];
      } else {
        dstl[buf][tid] = -1; idg[buf][tid] = 0.0f;
        ea[buf][tid * 3 + 0] = ea[buf][tid * 3 + 1] = ea[buf][tid * 3 + 2] = 0.0f;
      }
    }
  };

  stage(0, 0);
  for (int t = 0; t < MSG_TILES; ++t) {
    int buf = t & 1;
    wait_async0();            // wave's staging of tile t landed in LDS
    __syncthreads();          // all waves staged t; all done computing t-1
    if (t + 1 < MSG_TILES) stage(t + 1, buf ^ 1);  // overlaps compute below

    const f16* A = &Alds[buf][0];
    v8f acc = {};
#pragma unroll
    for (int kg = 0; kg < 8; kg += 4) {
      v16h a0 = load_a_frag_lds(A, MSG_AS, (kg + 0) * 32, lane);
      v16h a1 = load_a_frag_lds(A, MSG_AS, (kg + 1) * 32, lane);
      v16h a2 = load_a_frag_lds(A, MSG_AS, (kg + 2) * 32, lane);
      v16h a3 = load_a_frag_lds(A, MSG_AS, (kg + 3) * 32, lane);
      acc = wmma_step(a0, bfrag[kg + 0], acc);
      acc = wmma_step(a1, bfrag[kg + 1], acc);
      acc = wmma_step(a2, bfrag[kg + 2], acc);
      acc = wmma_step(a3, bfrag[kg + 3], acc);
    }

    int Mb = (lane >> 4) << 3;
    if (fullblk) {
      // Fast path: no per-element guards -> straight-line loads + atomics
#pragma unroll
      for (int v = 0; v < 8; ++v) {
        int M = Mb + v;
        int d = dstl[buf][M];
        float val = acc[v] + wb + ea[buf][M * 3 + 0] * w0 + ea[buf][M * 3 + 1] * w1
                  + ea[buf][M * 3 + 2] * w2;
        val = fmaxf(val, 0.0f) * idg[buf][M];
        atomicAdd(&agg[(size_t)d * HDIM + nlane], val);
      }
    } else {
#pragma unroll
      for (int v = 0; v < 8; ++v) {
        int M = Mb + v;
        int d = dstl[buf][M];
        if (d >= 0) {
          float val = acc[v] + wb + ea[buf][M * 3 + 0] * w0 + ea[buf][M * 3 + 1] * w1
                    + ea[buf][M * 3 + 2] * w2;
          val = fmaxf(val, 0.0f) * idg[buf][M];
          atomicAdd(&agg[(size_t)d * HDIM + nlane], val);
        }
      }
    }
  }
}

// ---------------------------------------------------------------------------
// Update MLP: h = relu([h | agg | xg[batch] | xbc[batch]] @ upd_W + b), K=512
// ---------------------------------------------------------------------------
constexpr int UPD_K     = 4 * HDIM;        // 512
constexpr int UPD_AS    = UPD_K + 8;       // 520 halves -> 1040B rows
constexpr int UPD_TILES = 4;               // 64 nodes per block

__global__ __launch_bounds__(256) void upd_kernel(
    const float* __restrict__ agg, const f16* __restrict__ xg_h, const f16* __restrict__ xbc_h,
    const int* __restrict__ batch, const f16* __restrict__ Wt,   // [128][512] fp16
    const float* __restrict__ upd_b,
    float* __restrict__ h, f16* __restrict__ h_half,
    float* __restrict__ xg_sum, int N) {
  __shared__ __align__(16) f16 Alds[16 * UPD_AS];
  __shared__ int nodel[16];
  __shared__ int gl[16];

  int tid  = threadIdx.x;
  int lane = tid & 31;
  int wave = tid >> 5;
  int n0   = wave * 16;

  v16h bfrag[16];
#pragma unroll
  for (int ks = 0; ks < 16; ++ks) bfrag[ks] = load_b_frag(Wt, UPD_K, n0, ks * 32, lane);

  int   nlane = n0 + (lane & 15);
  float bb = upd_b[nlane];

  int  base    = blockIdx.x * (16 * UPD_TILES);
  bool fullblk = (base + 16 * UPD_TILES) <= N;

  for (int t = 0; t < UPD_TILES; ++t) {
    int t0 = base + t * 16;
    {
      int m = tid >> 4, c = tid & 15;            // c: 32-half chunk (64B)
      int node = t0 + m;
      bool ok  = fullblk || node < N;
      int seg  = c >> 2;
      int part = (c & 3) * 32;
      f16* out = &Alds[m * UPD_AS + c * 32];
      if (!ok) {
        uint4 z{0, 0, 0, 0};
        *(uint4*)(out) = z; *(uint4*)(out + 8) = z;
        *(uint4*)(out + 16) = z; *(uint4*)(out + 24) = z;
      } else if (seg == 0) {
        const f16* p = h_half + (size_t)node * HDIM + part;
        *(uint4*)(out)      = *(const uint4*)(p);
        *(uint4*)(out + 8)  = *(const uint4*)(p + 8);
        *(uint4*)(out + 16) = *(const uint4*)(p + 16);
        *(uint4*)(out + 24) = *(const uint4*)(p + 24);
      } else if (seg == 1) {
        const float* p = agg + (size_t)node * HDIM + part;
#pragma unroll
        for (int jj = 0; jj < 32; ++jj) out[jj] = (f16)p[jj];
      } else {
        int g = batch[node];
        const f16* p = ((seg == 2) ? xg_h : xbc_h) + (size_t)g * HDIM + part;
        *(uint4*)(out)      = *(const uint4*)(p);
        *(uint4*)(out + 8)  = *(const uint4*)(p + 8);
        *(uint4*)(out + 16) = *(const uint4*)(p + 16);
        *(uint4*)(out + 24) = *(const uint4*)(p + 24);
      }
      if (c == 0) {
        nodel[m] = ok ? node : -1;
        gl[m]    = ok ? batch[node] : 0;
      }
    }
    __syncthreads();

    v8f acc = {};
#pragma unroll
    for (int kg = 0; kg < 16; kg += 4) {
      v16h a0 = load_a_frag_lds(Alds, UPD_AS, (kg + 0) * 32, lane);
      v16h a1 = load_a_frag_lds(Alds, UPD_AS, (kg + 1) * 32, lane);
      v16h a2 = load_a_frag_lds(Alds, UPD_AS, (kg + 2) * 32, lane);
      v16h a3 = load_a_frag_lds(Alds, UPD_AS, (kg + 3) * 32, lane);
      acc = wmma_step(a0, bfrag[kg + 0], acc);
      acc = wmma_step(a1, bfrag[kg + 1], acc);
      acc = wmma_step(a2, bfrag[kg + 2], acc);
      acc = wmma_step(a3, bfrag[kg + 3], acc);
    }

    int Mb = (lane >> 4) << 3;
    if (fullblk) {
#pragma unroll
      for (int v = 0; v < 8; ++v) {
        int M = Mb + v;
        int node = nodel[M];
        float val = fmaxf(acc[v] + bb, 0.0f);
        h[(size_t)node * HDIM + nlane]      = val;
        h_half[(size_t)node * HDIM + nlane] = (f16)val;
        atomicAdd(&xg_sum[gl[M] * HDIM + nlane], val);
      }
    } else {
#pragma unroll
      for (int v = 0; v < 8; ++v) {
        int M = Mb + v;
        int node = nodel[M];
        if (node >= 0) {
          float val = fmaxf(acc[v] + bb, 0.0f);
          h[(size_t)node * HDIM + nlane]      = val;
          h_half[(size_t)node * HDIM + nlane] = (f16)val;
          atomicAdd(&xg_sum[gl[M] * HDIM + nlane], val);
        }
      }
    }
    __syncthreads();
  }
}

// ---------------------------------------------------------------------------
// Decoder: out = relu(h @ W1 + b1) @ W2 + b2  (one node per block)
// ---------------------------------------------------------------------------
__global__ __launch_bounds__(HDIM) void dec_kernel(
    const float* __restrict__ h, const float* __restrict__ W1, const float* __restrict__ b1,
    const float* __restrict__ W2, const float* __restrict__ b2,
    float* __restrict__ out, int N) {
  int node = blockIdx.x;
  if (node >= N) return;
  __shared__ float hrow[HDIM];
  __shared__ float trow[HDIM];
  int j = threadIdx.x;
  hrow[j] = h[(size_t)node * HDIM + j];
  __syncthreads();
  float s = b1[j];
#pragma unroll 8
  for (int k = 0; k < HDIM; ++k) s = fmaf(hrow[k], W1[k * HDIM + j], s);
  trow[j] = fmaxf(s, 0.0f);
  __syncthreads();
  if (j < 3) {
    float o = b2[j];
#pragma unroll 8
    for (int k = 0; k < HDIM; ++k) o = fmaf(trow[k], W2[k * 3 + j], o);
    out[(size_t)node * 3 + j] = o;
  }
}

// ---------------------------------------------------------------------------
// Host launcher
// ---------------------------------------------------------------------------
extern "C" void kernel_launch(void* const* d_in, const int* in_sizes, int n_in,
                              void* d_out, int out_size, void* d_ws, size_t ws_size,
                              hipStream_t stream) {
  const float* x         = (const float*)d_in[0];
  const float* x_mask    = (const float*)d_in[1];
  const float* edge_attr = (const float*)d_in[2];
  const float* enc_W  = (const float*)d_in[4];
  const float* enc_b  = (const float*)d_in[5];
  const float* msg_W  = (const float*)d_in[6];
  const float* msg_b  = (const float*)d_in[7];
  const float* upd_W  = (const float*)d_in[8];
  const float* upd_b  = (const float*)d_in[9];
  const float* dec_W1 = (const float*)d_in[10];
  const float* dec_b1 = (const float*)d_in[11];
  const float* dec_W2 = (const float*)d_in[12];
  const float* dec_b2 = (const float*)d_in[13];
  const int* edge_index = (const int*)d_in[14];
  const int* batch      = (const int*)d_in[15];

  const int N = in_sizes[0] / 8;
  const int E = in_sizes[14] / 2;
  const int* src = edge_index;
  const int* dst = edge_index + E;

  char* cur = (char*)d_ws;
  auto alloc = [&](size_t bytes) -> void* {
    void* p = (void*)cur;
    cur += (bytes + 255) & ~(size_t)255;
    return p;
  };
  float* h        = (float*)alloc((size_t)N * HDIM * 4);
  f16*   h_half   = (f16*)  alloc((size_t)N * HDIM * 2);
  float* agg      = (float*)alloc((size_t)N * HDIM * 4);
  float* inv_deg  = (float*)alloc((size_t)N * 4);
  float* n_per_g  = (float*)alloc((size_t)GNUM * 4);
  float* bc_per_g = (float*)alloc((size_t)GNUM * 4);
  float* xg_sum   = (float*)alloc((size_t)GNUM * HDIM * 4);
  float* xbc_sum  = (float*)alloc((size_t)GNUM * HDIM * 4);
  f16*   xg_h     = (f16*)  alloc((size_t)GNUM * HDIM * 2);
  f16*   xbc_h    = (f16*)  alloc((size_t)GNUM * HDIM * 2);
  f16*   msg_Wt   = (f16*)  alloc((size_t)HDIM * MSG_K * 2);
  f16*   upd_Wt   = (f16*)  alloc((size_t)HDIM * UPD_K * 2);

  auto cdiv = [](int a, int b) { return (a + b - 1) / b; };

  zero_f32<<<2048, 256, 0, stream>>>(inv_deg, N);
  zero_f32<<<16, 256, 0, stream>>>(n_per_g, GNUM);
  zero_f32<<<16, 256, 0, stream>>>(bc_per_g, GNUM);
  zero_f32<<<16, 256, 0, stream>>>(xg_sum, GNUM * HDIM);
  zero_f32<<<16, 256, 0, stream>>>(xbc_sum, GNUM * HDIM);

  make_wt<<<cdiv(MSG_K * HDIM, 256), 256, 0, stream>>>(msg_W, msg_Wt, MSG_K);
  make_wt<<<cdiv(UPD_K * HDIM, 256), 256, 0, stream>>>(upd_W, upd_Wt, UPD_K);

  degree_kernel<<<cdiv(E, 256), 256, 0, stream>>>(dst, inv_deg, E);
  node_counts<<<cdiv(N, 256), 256, 0, stream>>>(batch, x_mask, n_per_g, bc_per_g, N);
  finalize_deg<<<cdiv(N, 256), 256, 0, stream>>>(inv_deg, N);

  enc_kernel<<<N, HDIM, 0, stream>>>(x, x_mask, enc_W, enc_b, batch, h, h_half,
                                     xg_sum, xbc_sum, N);
  finalize_graphs<<<cdiv(GNUM * HDIM, 256), 256, 0, stream>>>(xg_sum, xbc_sum, n_per_g,
                                                              bc_per_g, xg_h, xbc_h);

  for (int it = 0; it < REPEATS; ++it) {
    zero_f32<<<2048, 256, 0, stream>>>(agg, N * HDIM);
    msg_kernel<<<cdiv(E, 16 * MSG_TILES), 256, 0, stream>>>(
        h_half, edge_attr, src, dst, inv_deg, msg_Wt, msg_W, msg_b, agg, E);

    zero_f32<<<16, 256, 0, stream>>>(xg_sum, GNUM * HDIM);
    upd_kernel<<<cdiv(N, 16 * UPD_TILES), 256, 0, stream>>>(
        agg, xg_h, xbc_h, batch, upd_Wt, upd_b, h, h_half, xg_sum, N);

    finalize_xg<<<cdiv(GNUM * HDIM, 256), 256, 0, stream>>>(xg_sum, n_per_g, xg_h);
  }

  dec_kernel<<<N, HDIM, 0, stream>>>(h, dec_W1, dec_b1, dec_W2, dec_b2, (float*)d_out, N);
}
